// SelfAttentionLayer_55576876810790
// MI455X (gfx1250) — matrix operand order repaired and checked by reading, not compiled, not run on hardware
//
#include <hip/hip_runtime.h>
#include <hip/hip_bf16.h>

typedef __bf16 bf16_t;
typedef __attribute__((ext_vector_type(16))) __bf16 v16bf;
typedef __attribute__((ext_vector_type(8)))  float  v8f;
typedef unsigned int u32;
typedef __attribute__((ext_vector_type(4))) u32 u32x4;
typedef __attribute__((ext_vector_type(8))) int i32x8;
typedef __attribute__((ext_vector_type(4))) int i32x4;

#define EMBED 1024
#define NHEAD 16
#define HD    64
#define BATCH 4
#define SEQ   2048
#define ROWS  (BATCH*SEQ)   /* 8192 */
#define C3    (3*EMBED)     /* 3072 */
#define QKC   2048          /* q+k columns */

#if __has_builtin(__builtin_amdgcn_tensor_load_to_lds) && __has_builtin(__builtin_amdgcn_s_wait_tensorcnt)
#define USE_TDM 1
#else
#define USE_TDM 0
#endif

union FragBF { v16bf v; uint4 q[2]; };
union Pack8  { bf16_t h[8]; uint4 q; };

__device__ __forceinline__ u32 lds_addr32(const void* p) {
    // generic address of LDS: low 32 bits == group-segment byte offset
    return (u32)(unsigned long long)p;
}

#if USE_TDM
// 2-D bf16 tile DMA: tile tw x th elements, row stride `stride` elements.
__device__ __forceinline__ void tdm_load_2d_bf16(u32 lds, const bf16_t* g,
                                                 u32 tw, u32 th, u32 stride)
{
    unsigned long long ga = (unsigned long long)g;
    u32x4 g0;
    g0.x = 1u;                                              // count=1, user desc
    g0.y = lds;                                             // lds_addr (bytes)
    g0.z = (u32)ga;                                         // global_addr[31:0]
    g0.w = (u32)((ga >> 32) & 0x01FFFFFFu) | (2u << 30);    // addr[56:32] | type=2
    i32x8 g1;
    g1[0] = 1 << 16;                                        // data_size=1 (2B); mask=0
    g1[1] = (int)((tw & 0xFFFFu) << 16);                    // tensor_dim0[15:0]
    g1[2] = (int)((tw >> 16) | ((th & 0xFFFFu) << 16));     // dim0 hi | dim1 lo
    g1[3] = (int)((th >> 16) | (tw << 16));                 // dim1 hi | tile_dim0
    g1[4] = (int)(th & 0xFFFFu);                            // tile_dim1 (tile_dim2=0)
    g1[5] = (int)stride;                                    // tensor_dim0_stride lo32
    g1[6] = 0;
    g1[7] = 0;
    i32x4 z4 = {0,0,0,0};
    i32x8 z8 = {0,0,0,0,0,0,0,0};
    // 6-arg form (clang-23 / therock-10.0 headers): extra int32x8 before cpol
    __builtin_amdgcn_tensor_load_to_lds(g0, g1, z4, z4, z8, 0);
}
#endif

// ---------------------------------------------------------------- converts
__global__ void cvt_f32_to_bf16(const float* __restrict__ in,
                                bf16_t* __restrict__ out, int n)
{
    int i = blockIdx.x * blockDim.x + threadIdx.x;
    int stride = gridDim.x * blockDim.x;
    for (; i < n; i += stride) out[i] = (bf16_t)in[i];
}

// in[K][N] f32 -> out[N][K] bf16  (one-time weight transpose)
__global__ void cvt_transpose_f32_to_bf16(const float* __restrict__ in,
                                          bf16_t* __restrict__ out, int K, int N)
{
    int i = blockIdx.x * blockDim.x + threadIdx.x;
    int total = K * N;
    int stride = gridDim.x * blockDim.x;
    for (; i < total; i += stride) {
        int k = i / N, n = i - k * N;
        out[(size_t)n * K + k] = (bf16_t)in[i];
    }
}

// ---------------------------------------------------------------- GEMM
// C[M,N] = A[M,K] * Bt[N,K]^T, bf16 in / f32 accum. Block tile 128x64, BK=32.
// MODE 0: f32 row-major to outF.
// MODE 2: qkv split: cols < 2048 -> outQK row-major [row][2048] bf16;
//         cols >= 2048 -> outVT transposed [b][h][d][t] bf16 (packed b128 store).
template<int MODE>
__global__ __launch_bounds__(256) void gemm_wmma_bf16(
    const bf16_t* __restrict__ A, const bf16_t* __restrict__ Bt,
    float* __restrict__ outF, bf16_t* __restrict__ outQK, bf16_t* __restrict__ outVT,
    int M, int N, int K)
{
    __shared__ bf16_t As[2][128 * 32];   // [m][k]
    __shared__ bf16_t Bs[2][64 * 32];    // [n][k]

    const int tid  = threadIdx.x;
    const int wid  = tid >> 5;
    const int lane = tid & 31;
    const int lh   = lane >> 4;
    const int l16  = lane & 15;
    const int ntn  = N >> 6;
    const int bm   = blockIdx.x / ntn;
    const int bn   = blockIdx.x % ntn;
    const int wM   = wid & 3;
    const int wN   = wid >> 2;

    const bf16_t* Abase = A  + (size_t)bm * 128 * K;
    const bf16_t* Bbase = Bt + (size_t)bn * 64  * K;

    v8f vz = {0.f,0.f,0.f,0.f,0.f,0.f,0.f,0.f};
    v8f acc[2][2];
    acc[0][0]=vz; acc[0][1]=vz; acc[1][0]=vz; acc[1][1]=vz;

    const int nIter = K >> 5;

#if USE_TDM
    if (wid == 0) {
        tdm_load_2d_bf16(lds_addr32(&As[0][0]), Abase, 32, 128, (u32)K);
        tdm_load_2d_bf16(lds_addr32(&Bs[0][0]), Bbase, 32, 64,  (u32)K);
    }
#else
    const int arow = tid >> 1, acol = (tid & 1) * 16;   // A: 128 rows x 32
    const int brow = tid >> 2, bcol = (tid & 3) * 8;    // B: 64 rows x 32
    {
        const uint4* s = (const uint4*)(Abase + (size_t)arow * K + acol);
        uint4* d = (uint4*)(&As[0][arow*32 + acol]);
        d[0] = s[0]; d[1] = s[1];
        *(uint4*)(&Bs[0][brow*32 + bcol]) =
            *(const uint4*)(Bbase + (size_t)brow * K + bcol);
    }
#endif

    for (int it = 0; it < nIter; ++it) {
        const int cur = it & 1;
        const int nxt = cur ^ 1;
#if USE_TDM
        if (wid == 0) {
            if (it + 1 < nIter) {
                const int k1 = (it + 1) << 5;
                tdm_load_2d_bf16(lds_addr32(&As[nxt][0]), Abase + k1, 32, 128, (u32)K);
                tdm_load_2d_bf16(lds_addr32(&Bs[nxt][0]), Bbase + k1, 32, 64,  (u32)K);
                __builtin_amdgcn_s_wait_tensorcnt(2);   // current buffer complete
            } else {
                __builtin_amdgcn_s_wait_tensorcnt(0);
            }
        }
#else
        if (it + 1 < nIter) {
            const int k1 = (it + 1) << 5;
            const uint4* s = (const uint4*)(Abase + (size_t)arow * K + k1 + acol);
            uint4* d = (uint4*)(&As[nxt][arow*32 + acol]);
            d[0] = s[0]; d[1] = s[1];
            *(uint4*)(&Bs[nxt][brow*32 + bcol]) =
                *(const uint4*)(Bbase + (size_t)brow * K + k1 + bcol);
        }
#endif
        __syncthreads();

        FragBF af[2], bfr[2];
        #pragma unroll
        for (int i = 0; i < 2; ++i) {
            const int r = wM*32 + i*16 + l16;
            af[i].q[0] = *(const uint4*)(&As[cur][r*32 + lh*8]);
            af[i].q[1] = *(const uint4*)(&As[cur][r*32 + 16 + lh*8]);
        }
        #pragma unroll
        for (int j = 0; j < 2; ++j) {
            const int c = wN*32 + j*16 + l16;
            bfr[j].q[0] = *(const uint4*)(&Bs[cur][c*32 + lh*8]);
            bfr[j].q[1] = *(const uint4*)(&Bs[cur][c*32 + 16 + lh*8]);
        }
        #pragma unroll
        for (int i = 0; i < 2; ++i)
            #pragma unroll
            for (int j = 0; j < 2; ++j)
                acc[i][j] = __builtin_amdgcn_wmma_f32_16x16x32_bf16(
                    false, af[i].v, false, bfr[j].v, (short)0, acc[i][j], false, false);
        __syncthreads();
    }

    #pragma unroll
    for (int i = 0; i < 2; ++i)
      #pragma unroll
      for (int j = 0; j < 2; ++j) {
        const int rowbase = bm*128 + wM*32 + i*16 + lh*8;
        const int col     = bn*64  + wN*32 + j*16 + l16;
        if (MODE == 0) {
            #pragma unroll
            for (int r = 0; r < 8; ++r)
                outF[(size_t)(rowbase + r) * N + col] = acc[i][j][r];
        } else {
            if (col < QKC) {                       // Q/K: row-major bf16
                #pragma unroll
                for (int r = 0; r < 8; ++r)
                    outQK[(size_t)(rowbase + r) * QKC + col] = (bf16_t)acc[i][j][r];
            } else {                               // V: transposed [b][h][d][t]
                const int v  = col - QKC;
                const int h  = v >> 6;
                const int d  = v & 63;
                const int b  = rowbase >> 11;      // 2048 rows per batch
                const int t0 = rowbase & 2047;     // multiple of 8
                Pack8 p;
                #pragma unroll
                for (int r = 0; r < 8; ++r) p.h[r] = (bf16_t)acc[i][j][r];
                *(uint4*)(outVT + (((size_t)b*NHEAD + h)*HD + d)*SEQ + t0) = p.q;
            }
        }
      }
}

// ---------------------------------------------------------------- attention
// One wave per (b, h, 16-query tile). Q/K from qk[row][2048]; V from vT[b][h][d][t].
__global__ __launch_bounds__(256) void flash_attn_wmma(
    const bf16_t* __restrict__ qk, const bf16_t* __restrict__ vT,
    bf16_t* __restrict__ y)
{
    __shared__ bf16_t Ps[8][16 * 32];    // per-wave P staging (C-layout -> A-layout)

    const int tid  = threadIdx.x;
    const int wid  = tid >> 5;
    const int lane = tid & 31;
    const int lh   = lane >> 4;
    const int l16  = lane & 15;

    const int gw = blockIdx.x * 8 + wid;
    const int qt = gw & 127;
    const int bh = gw >> 7;
    const int b  = bh >> 4;
    const int h  = bh & 15;
    const int q0 = qt * 16;

    const size_t rs = QKC;
    const bf16_t* qb = qk + (size_t)b*SEQ*rs + h*HD;
    const bf16_t* kb = qb + EMBED;
    const bf16_t* vb = vT + (size_t)bh * HD * SEQ;   // [d][t]

    FragBF qf[2];
    #pragma unroll
    for (int c = 0; c < 2; ++c) {
        const bf16_t* qr = qb + (size_t)(q0 + l16)*rs + c*32;
        qf[c].q[0] = *(const uint4*)(qr + lh*8);
        qf[c].q[1] = *(const uint4*)(qr + 16 + lh*8);
    }

    v16bf ones;
    #pragma unroll
    for (int e = 0; e < 16; ++e) ones[e] = (bf16_t)1.0f;

    v8f vz = {0.f,0.f,0.f,0.f,0.f,0.f,0.f,0.f};
    float m[8], l[8];
    v8f O[4];
    #pragma unroll
    for (int r = 0; r < 8; ++r) { m[r] = -1e30f; l[r] = 0.f; }
    #pragma unroll
    for (int d = 0; d < 4; ++d) O[d] = vz;

    bf16_t* myP = Ps[wid];
    const float scl = 0.125f;            // 1/sqrt(64)
    const int jmax = (q0 + 15) >> 5;

    for (int j = 0; j <= jmax; ++j) {
        const int kc = j << 5;

        // ---- S = Q @ K^T (two 16x16 subtiles), K frags direct from global
        v8f S[2];
        #pragma unroll
        for (int t = 0; t < 2; ++t) {
            v8f s = vz;
            #pragma unroll
            for (int c = 0; c < 2; ++c) {
                FragBF kf;
                const bf16_t* kr = kb + (size_t)(kc + t*16 + l16)*rs + c*32 + lh*8;
                kf.q[0] = ((const uint4*)kr)[0];
                kf.q[1] = ((const uint4*)kr)[2];
                s = __builtin_amdgcn_wmma_f32_16x16x32_bf16(
                        false, qf[c].v, false, kf.v, (short)0, s, false, false);
            }
            S[t] = s;
        }

        // ---- scale + causal mask (diagonal chunk only)
        #pragma unroll
        for (int t = 0; t < 2; ++t)
            #pragma unroll
            for (int r = 0; r < 8; ++r) S[t][r] *= scl;
        if (kc + 31 > q0) {
            #pragma unroll
            for (int t = 0; t < 2; ++t)
                #pragma unroll
                for (int r = 0; r < 8; ++r) {
                    const int qi = q0 + r + lh*8;
                    const int kj = kc + t*16 + l16;
                    if (kj > qi) S[t][r] = -1e30f;
                }
        }

        // ---- row max (shuffle within 16-lane halves) + rescale factor
        float sc[8];
        #pragma unroll
        for (int r = 0; r < 8; ++r) {
            float v = fmaxf(S[0][r], S[1][r]);
            v = fmaxf(v, __shfl_xor(v, 1, 32));
            v = fmaxf(v, __shfl_xor(v, 2, 32));
            v = fmaxf(v, __shfl_xor(v, 4, 32));
            v = fmaxf(v, __shfl_xor(v, 8, 32));
            const float mn = fmaxf(m[r], v);
            sc[r] = __expf(m[r] - mn);
            m[r] = mn;
        }

        // ---- P = exp(S - m) -> LDS (layout change), reload as A-fragment
        #pragma unroll
        for (int r = 0; r < 8; ++r) {
            const int row = r + lh*8;
            myP[row*32 + l16]      = (bf16_t)__expf(S[0][r] - m[r]);
            myP[row*32 + 16 + l16] = (bf16_t)__expf(S[1][r] - m[r]);
        }
        FragBF pf;
        pf.q[0] = *(const uint4*)(&myP[l16*32 + lh*8]);
        pf.q[1] = *(const uint4*)(&myP[l16*32 + 16 + lh*8]);

        // ---- row sums via WMMA: P @ ones (replaces 32 shuffles)
        v8f ls = __builtin_amdgcn_wmma_f32_16x16x32_bf16(
                    false, pf.v, false, ones, (short)0, vz, false, false);
        #pragma unroll
        for (int r = 0; r < 8; ++r) l[r] = l[r]*sc[r] + ls[r];

        // ---- O = O*diag(sc) + P @ V, V frags straight from vT[d][t]
        #pragma unroll
        for (int dn = 0; dn < 4; ++dn) {
            FragBF vf;
            const bf16_t* vr = vb + (size_t)(dn*16 + l16)*SEQ + kc + lh*8;
            vf.q[0] = ((const uint4*)vr)[0];
            vf.q[1] = ((const uint4*)vr)[2];
            v8f o = O[dn];
            #pragma unroll
            for (int r = 0; r < 8; ++r) o[r] *= sc[r];
            O[dn] = __builtin_amdgcn_wmma_f32_16x16x32_bf16(
                        false, pf.v, false, vf.v, (short)0, o, false, false);
        }
    }

    float inv[8];
    #pragma unroll
    for (int r = 0; r < 8; ++r) inv[r] = 1.0f / l[r];
    #pragma unroll
    for (int dn = 0; dn < 4; ++dn)
        #pragma unroll
        for (int r = 0; r < 8; ++r) {
            const int row = r + lh*8;
            y[(size_t)(b*SEQ + q0 + row)*EMBED + h*HD + dn*16 + l16] =
                (bf16_t)(O[dn][r] * inv[r]);
        }
}

// ---------------------------------------------------------------- launcher
extern "C" void kernel_launch(void* const* d_in, const int* in_sizes, int n_in,
                              void* d_out, int out_size, void* d_ws, size_t ws_size,
                              hipStream_t stream)
{
    (void)in_sizes; (void)n_in; (void)out_size; (void)ws_size;
    const float* x  = (const float*)d_in[0];
    const float* wa = (const float*)d_in[1];
    const float* wp = (const float*)d_in[2];
    float* out = (float*)d_out;
    char* ws = (char*)d_ws;

    const size_t nx  = (size_t)ROWS * EMBED;

    bf16_t* xb   = (bf16_t*)ws;  ws += nx * 2;                       // 16 MB
    bf16_t* wabt = (bf16_t*)ws;  ws += (size_t)C3 * EMBED * 2;       //  6 MB [3072][1024]
    bf16_t* wpbt = (bf16_t*)ws;  ws += (size_t)EMBED * EMBED * 2;    //  2 MB [1024][1024]
    bf16_t* qkb  = (bf16_t*)ws;  ws += (size_t)ROWS * QKC * 2;       // 32 MB [row][2048]
    bf16_t* vTb  = (bf16_t*)ws;  ws += (size_t)BATCH*NHEAD*HD*SEQ*2; // 16 MB [b][h][d][t]
    bf16_t* yb   = (bf16_t*)ws;                                      // 16 MB

    cvt_f32_to_bf16<<<2048, 256, 0, stream>>>(x, xb, (int)nx);
    cvt_transpose_f32_to_bf16<<<1024, 256, 0, stream>>>(wa, wabt, EMBED, C3);
    cvt_transpose_f32_to_bf16<<<512,  256, 0, stream>>>(wp, wpbt, EMBED, EMBED);

    // qkv = x @ w_attn : 8192 x 3072 x 1024 (Q/K row-major, V transposed)
    gemm_wmma_bf16<2><<<dim3((ROWS/128)*(C3/64)), 256, 0, stream>>>(
        xb, wabt, nullptr, qkb, vTb, ROWS, C3, EMBED);

    // attention: 64 (b,h) * 128 q-tiles = 8192 waves
    flash_attn_wmma<<<dim3(1024), 256, 0, stream>>>(qkb, vTb, yb);

    // out = y @ w_proj : 8192 x 1024 x 1024, f32 out
    gemm_wmma_bf16<0><<<dim3((ROWS/128)*(EMBED/64)), 256, 0, stream>>>(
        yb, wpbt, out, nullptr, nullptr, ROWS, EMBED, EMBED);
}
